// AttentionBlock_5909874999657
// MI455X (gfx1250) — compile-verified, matrix operand order checked by reading, not run on hardware
//
#include <hip/hip_runtime.h>

#define C_DIM 512
#define N_DIM 4096
#define B_DIM 4
#define CC (C_DIM * C_DIM)   // 262144 = 2^18

typedef __attribute__((ext_vector_type(16))) _Float16 v16h;
typedef __attribute__((ext_vector_type(8)))  float    v8f;

__device__ __forceinline__ int lane_id() { return (int)(threadIdx.x & 31); }

// CDNA5 16x16x32 f16 WMMA operand layout (ISA 7.12.2): lane holds row = lane&15,
// K = {kb..kb+7, kb+16..kb+23} with kb = (lane&16)?8:0  ->  two contiguous
// 16-byte chunks per lane when storage is row-major in K.
union FragU {
  v16h v;
  struct { int4 lo; int4 hi; } q;
};

__device__ __forceinline__ v16h load_frag_rm(const _Float16* base, int stride) {
  const int lane = lane_id();
  const int row = lane & 15;
  const int kb = (lane & 16) ? 8 : 0;
  const _Float16* p = base + row * stride + kb;
  FragU u;
  u.q.lo = *(const int4*)(p);
  u.q.hi = *(const int4*)(p + 16);
  return u.v;
}

__device__ __forceinline__ v8f wmma16(v16h a, v16h b, v8f c) {
  return __builtin_amdgcn_wmma_f32_16x16x32_f16(false, a, false, b, (short)0, c,
                                                false, false);
}

__device__ __forceinline__ v8f zero8() {
  v8f z = {0.f, 0.f, 0.f, 0.f, 0.f, 0.f, 0.f, 0.f};
  return z;
}

// ---------------------------------------------------------------- kernel 0
__global__ void cvt_w_kernel(const float* __restrict__ wq, const float* __restrict__ wk,
                             const float* __restrict__ wv, const float* __restrict__ wp,
                             _Float16* __restrict__ dst) {
  int i = blockIdx.x * blockDim.x + threadIdx.x;
  int which = i >> 18;
  int j = i & (CC - 1);
  const float* src = (which == 0) ? wq : (which == 1) ? wk : (which == 2) ? wv : wp;
  dst[i] = (_Float16)src[j];
}

// ---------------------------------------------------------------- kernel 1a
__global__ void in_stats_kernel(const float* __restrict__ x, float2* __restrict__ nstats) {
  const int plane = blockIdx.x;  // b*C + c
  const float* xp = x + (size_t)plane * N_DIM;
  __shared__ float s1[256], s2[256];
  float a = 0.f, b2 = 0.f;
  for (int n = threadIdx.x; n < N_DIM; n += 256) {
    float v = xp[n];
    a += v;
    b2 += v * v;
  }
  s1[threadIdx.x] = a;
  s2[threadIdx.x] = b2;
  __syncthreads();
  for (int off = 128; off > 0; off >>= 1) {
    if ((int)threadIdx.x < off) {
      s1[threadIdx.x] += s1[threadIdx.x + off];
      s2[threadIdx.x] += s2[threadIdx.x + off];
    }
    __syncthreads();
  }
  if (threadIdx.x == 0) {
    float m = s1[0] * (1.f / N_DIM);
    float var = s2[0] * (1.f / N_DIM) - m * m;
    nstats[plane] = make_float2(m, rsqrtf(var + 1e-5f));
  }
}

// ---------------------------------------------------------------- kernel 1b
__global__ void norm_t_kernel(const float* __restrict__ x, const float2* __restrict__ nstats,
                              _Float16* __restrict__ hnT) {
  const int n0 = blockIdx.x * 64, c0 = blockIdx.y * 64, b = blockIdx.z;
  __shared__ _Float16 T[64 * 72];  // [n][c]
  const int t = threadIdx.x;
  {
    const int c = t >> 2;
    const int nn = (t & 3) * 16;
    const float2 st = nstats[(size_t)b * C_DIM + c0 + c];
    const float* xp = x + ((size_t)b * C_DIM + c0 + c) * N_DIM + n0 + nn;
#pragma unroll
    for (int e = 0; e < 16; e += 4) {
      float4 v4 = *(const float4*)&xp[e];
      T[(nn + e + 0) * 72 + c] = (_Float16)((v4.x - st.x) * st.y);
      T[(nn + e + 1) * 72 + c] = (_Float16)((v4.y - st.x) * st.y);
      T[(nn + e + 2) * 72 + c] = (_Float16)((v4.z - st.x) * st.y);
      T[(nn + e + 3) * 72 + c] = (_Float16)((v4.w - st.x) * st.y);
    }
  }
  __syncthreads();
  {
    const int n = t >> 2;
    const int cc = (t & 3) * 16;
    _Float16* dst = hnT + ((size_t)b * N_DIM + n0 + n) * C_DIM + c0 + cc;
    *(int4*)&dst[0] = *(const int4*)&T[n * 72 + cc];
    *(int4*)&dst[8] = *(const int4*)&T[n * 72 + cc + 8];
  }
}

// ---------------------------------------------------------------- kernel 2
// rm x rm GEMM: D[m][nl] = sum_k A[m][k] * B[nl][k], K = 512. Register-prefetch
// double buffering: stores of tile k happen at loop top, loads of tile k+1 are
// issued before the WMMA stage so their wait hides behind compute.
// Block tile 128(M) x 256(N); 8 waves as 2x4 of 64x64 (16 WMMA / wave / step).
template <int MODE>
__global__ __launch_bounds__(256, 1) void gemm_rr_kernel(
    const _Float16* __restrict__ w16, const _Float16* __restrict__ srcT,
    _Float16* __restrict__ d0, _Float16* __restrict__ d1,
    const float* __restrict__ bias0, const float* __restrict__ bias1,
    const float* __restrict__ xres, float* __restrict__ outf) {
  const size_t plane = (size_t)C_DIM * N_DIM;
  const int m0 = blockIdx.y * 128;
  const int n0g = blockIdx.x * 256;
  int b, sel = 0;
  const _Float16 *A, *B;
  const float* bias;
  _Float16* dst16 = nullptr;
  float scale = 1.f;
  if (MODE == 0) {
    sel = blockIdx.z & 1;
    b = blockIdx.z >> 1;
    A = srcT + (size_t)b * plane;             // hnT [n][c]
    B = w16 + (size_t)sel * CC;               // wq/wk [o][c]
    bias = sel ? bias1 : bias0;
    dst16 = sel ? d1 : d0;
    if (!sel) scale = 0.044194173824159216f;  // 1/sqrt(512)
  } else if (MODE == 2) {
    b = blockIdx.z;
    A = w16 + (size_t)2 * CC;                 // wv [c_out][c]
    B = srcT + (size_t)b * plane;             // hnT [n][c]
    bias = bias0;
    dst16 = d0;
  } else {
    b = blockIdx.z;
    A = w16 + (size_t)3 * CC;                 // wp [c_out][c]
    B = srcT + (size_t)b * plane;             // oT [n][c]
    bias = bias0;
  }

  __shared__ _Float16 As[128 * 40];  // [m][k] rm, pad to 40
  __shared__ _Float16 Bs[256 * 40];  // [nl][k] rm
  const int t = threadIdx.x;
  const int wave = t >> 5;
  const int wm = wave >> 2;  // 0..1
  const int wn = wave & 3;   // 0..3
  const int srow = t >> 2;          // 0..63
  const int skk = (t & 3) * 8;      // 0,8,16,24

  // per-thread staging source pointers (advance by 32 each K step)
  const _Float16* ga0 = &A[(size_t)(m0 + srow) * C_DIM + skk];
  const _Float16* ga1 = &A[(size_t)(m0 + srow + 64) * C_DIM + skk];
  const _Float16* gb0 = &B[(size_t)(n0g + srow) * C_DIM + skk];
  const _Float16* gb1 = &B[(size_t)(n0g + srow + 64) * C_DIM + skk];
  const _Float16* gb2 = &B[(size_t)(n0g + srow + 128) * C_DIM + skk];
  const _Float16* gb3 = &B[(size_t)(n0g + srow + 192) * C_DIM + skk];

  v8f acc[4][4];
#pragma unroll
  for (int i = 0; i < 4; ++i)
#pragma unroll
    for (int j = 0; j < 4; ++j) acc[i][j] = zero8();

  // prefetch tile 0
  int4 pa0 = *(const int4*)ga0, pa1 = *(const int4*)ga1;
  int4 pb0 = *(const int4*)gb0, pb1 = *(const int4*)gb1;
  int4 pb2 = *(const int4*)gb2, pb3 = *(const int4*)gb3;

#pragma unroll 1
  for (int k0 = 0; k0 < C_DIM; k0 += 32) {
    // commit prefetched tile to LDS
    *(int4*)&As[(srow) * 40 + skk] = pa0;
    *(int4*)&As[(srow + 64) * 40 + skk] = pa1;
    *(int4*)&Bs[(srow) * 40 + skk] = pb0;
    *(int4*)&Bs[(srow + 64) * 40 + skk] = pb1;
    *(int4*)&Bs[(srow + 128) * 40 + skk] = pb2;
    *(int4*)&Bs[(srow + 192) * 40 + skk] = pb3;
    __syncthreads();
    // issue next tile's global loads (wait lands at next iteration's stores)
    if (k0 + 32 < C_DIM) {
      pa0 = *(const int4*)(ga0 + k0 + 32);
      pa1 = *(const int4*)(ga1 + k0 + 32);
      pb0 = *(const int4*)(gb0 + k0 + 32);
      pb1 = *(const int4*)(gb1 + k0 + 32);
      pb2 = *(const int4*)(gb2 + k0 + 32);
      pb3 = *(const int4*)(gb3 + k0 + 32);
    }
    v16h af[4], bf[4];
#pragma unroll
    for (int i = 0; i < 4; ++i) af[i] = load_frag_rm(&As[(wm * 64 + i * 16) * 40], 40);
#pragma unroll
    for (int j = 0; j < 4; ++j) bf[j] = load_frag_rm(&Bs[(wn * 64 + j * 16) * 40], 40);
#pragma unroll
    for (int i = 0; i < 4; ++i)
#pragma unroll
      for (int j = 0; j < 4; ++j) acc[i][j] = wmma16(af[i], bf[j], acc[i][j]);
    __syncthreads();
  }

  const int lane = lane_id();
  const int ln = lane & 15;
  const int mh = (lane >> 4) * 8;
#pragma unroll
  for (int i = 0; i < 4; ++i)
#pragma unroll
    for (int j = 0; j < 4; ++j) {
      const int mb = m0 + wm * 64 + i * 16 + mh;
      const int nl = n0g + wn * 64 + j * 16 + ln;
#pragma unroll
      for (int r = 0; r < 8; ++r) {
        int m = mb + r;
        float v = acc[i][j][r];
        if (MODE == 0) {
          v = (v + bias[nl]) * scale;
          dst16[((size_t)b * N_DIM + m) * C_DIM + nl] = (_Float16)v;
        } else if (MODE == 2) {
          v += bias[m];
          dst16[((size_t)b * C_DIM + m) * N_DIM + nl] = (_Float16)v;
        } else {
          size_t idx = ((size_t)b * C_DIM + m) * N_DIM + nl;
          outf[idx] = xres[idx] + v + bias[m];
        }
      }
    }
}

// Shared device helper: one S tile (64 x 128) at (i0, j0) with register-prefetch
// staging. Result left in acc[2][2] per wave (2x4 wave grid of 32x32 subtiles).
__device__ __forceinline__ void s_tile_gemm(const _Float16* __restrict__ qb,
                                            const _Float16* __restrict__ kb,
                                            _Float16* Qs, _Float16* Ks, int i0, int j0,
                                            int t, int wi, int wj, v8f acc[2][2]) {
  const int srow = t >> 2;
  const int skk = (t & 3) * 8;
  const _Float16* gq = &qb[(size_t)(i0 + srow) * C_DIM + skk];
  const _Float16* gk0 = &kb[(size_t)(j0 + srow) * C_DIM + skk];
  const _Float16* gk1 = &kb[(size_t)(j0 + srow + 64) * C_DIM + skk];

  int4 pq = *(const int4*)gq;
  int4 pk0 = *(const int4*)gk0;
  int4 pk1 = *(const int4*)gk1;

#pragma unroll 1
  for (int c0 = 0; c0 < C_DIM; c0 += 32) {
    *(int4*)&Qs[srow * 40 + skk] = pq;
    *(int4*)&Ks[srow * 40 + skk] = pk0;
    *(int4*)&Ks[(srow + 64) * 40 + skk] = pk1;
    __syncthreads();
    if (c0 + 32 < C_DIM) {
      pq = *(const int4*)(gq + c0 + 32);
      pk0 = *(const int4*)(gk0 + c0 + 32);
      pk1 = *(const int4*)(gk1 + c0 + 32);
    }
    v16h af[2], bf[2];
#pragma unroll
    for (int i = 0; i < 2; ++i) af[i] = load_frag_rm(&Qs[(wi * 32 + i * 16) * 40], 40);
#pragma unroll
    for (int j = 0; j < 2; ++j) bf[j] = load_frag_rm(&Ks[(wj * 32 + j * 16) * 40], 40);
#pragma unroll
    for (int i = 0; i < 2; ++i)
#pragma unroll
      for (int j = 0; j < 2; ++j) acc[i][j] = wmma16(af[i], bf[j], acc[i][j]);
    __syncthreads();
  }
}

// ---------------------------------------------------------------- kernel 3
__global__ __launch_bounds__(256, 1) void attn_stats_kernel(
    const _Float16* __restrict__ qT, const _Float16* __restrict__ kT,
    float2* __restrict__ stats) {
  const int i0 = blockIdx.x * 64;
  const int b = blockIdx.y;
  __shared__ _Float16 Qs[64 * 40];   // [i][c]
  __shared__ _Float16 Ks[128 * 40];  // [j][c]
  __shared__ float Sb[64 * 132];     // [i][j]
  const int t = threadIdx.x;
  const int wave = t >> 5;
  const int wi = wave >> 2;  // 0..1
  const int wj = wave & 3;   // 0..3
  const _Float16* qb = qT + (size_t)b * C_DIM * N_DIM;
  const _Float16* kb = kT + (size_t)b * C_DIM * N_DIM;
  float m_run = -1e30f, l_run = 0.f;

#pragma unroll 1
  for (int j0 = 0; j0 < N_DIM; j0 += 128) {
    v8f acc[2][2];
#pragma unroll
    for (int i = 0; i < 2; ++i)
#pragma unroll
      for (int j = 0; j < 2; ++j) acc[i][j] = zero8();

    s_tile_gemm(qb, kb, Qs, Ks, i0, j0, t, wi, wj, acc);

    const int lane = lane_id();
    const int ln = lane & 15;
    const int mh = (lane >> 4) * 8;
#pragma unroll
    for (int i = 0; i < 2; ++i)
#pragma unroll
      for (int j = 0; j < 2; ++j)
#pragma unroll
        for (int r = 0; r < 8; ++r)
          Sb[(wi * 32 + i * 16 + mh + r) * 132 + wj * 32 + j * 16 + ln] = acc[i][j][r];
    __syncthreads();

    if (t < 64) {
      const float* row = &Sb[t * 132];
      float mt = -1e30f;
      for (int j = 0; j < 128; ++j) mt = fmaxf(mt, row[j]);
      float nm = fmaxf(m_run, mt);
      float s = 0.f;
      for (int j = 0; j < 128; ++j) s += __expf(row[j] - nm);
      l_run = l_run * __expf(m_run - nm) + s;
      m_run = nm;
    }
    __syncthreads();
  }
  if (t < 64) stats[(size_t)b * N_DIM + i0 + t] = make_float2(m_run, l_run);
}

// ---------------------------------------------------------------- kernel 4
union SPunion {
  float S[64 * 132];
  _Float16 P[64 * 136];
};

__global__ __launch_bounds__(256, 1) void attn_out_kernel(
    const _Float16* __restrict__ qT, const _Float16* __restrict__ kT,
    const _Float16* __restrict__ v16, const float2* __restrict__ stats,
    _Float16* __restrict__ oT) {
  const int i0 = blockIdx.x * 64;
  const int b = blockIdx.y;
  __shared__ _Float16 Qs[64 * 40];
  __shared__ _Float16 Ks[128 * 40];
  __shared__ SPunion u;
  __shared__ float mrow[64], linv[64];
  const int t = threadIdx.x;
  const int wave = t >> 5;
  const int lane = t & 31;
  const int wi = wave >> 2;
  const int wj = wave & 3;
  const _Float16* qb = qT + (size_t)b * C_DIM * N_DIM;
  const _Float16* kb = kT + (size_t)b * C_DIM * N_DIM;
  const _Float16* vb = v16 + (size_t)b * C_DIM * N_DIM;

  if (t < 64) {
    float2 st = stats[(size_t)b * N_DIM + i0 + t];
    mrow[t] = st.x;
    linv[t] = 1.f / st.y;
  }
  __syncthreads();

  // Wave owns channels [wave*64, wave*64+64), all 64 queries: D(m=i, n=c) 4x4 tiles
  v8f oacc[4][4];
#pragma unroll
  for (int ii = 0; ii < 4; ++ii)
#pragma unroll
    for (int ci = 0; ci < 4; ++ci) oacc[ii][ci] = zero8();

#pragma unroll 1
  for (int j0 = 0; j0 < N_DIM; j0 += 128) {
    // ---- S tile (64 x 128), cooperative
    v8f acc[2][2];
#pragma unroll
    for (int i = 0; i < 2; ++i)
#pragma unroll
      for (int j = 0; j < 2; ++j) acc[i][j] = zero8();

    s_tile_gemm(qb, kb, Qs, Ks, i0, j0, t, wi, wj, acc);

    {
      const int ln = lane & 15;
      const int mh = (lane >> 4) * 8;
#pragma unroll
      for (int i = 0; i < 2; ++i)
#pragma unroll
        for (int j = 0; j < 2; ++j)
#pragma unroll
          for (int r = 0; r < 8; ++r)
            u.S[(wi * 32 + i * 16 + mh + r) * 132 + wj * 32 + j * 16 + ln] = acc[i][j][r];
    }
    __syncthreads();

    // ---- P = exp(S - m) -> f16 (union overlay: regs, sync, write)
    float tmp[32];
#pragma unroll
    for (int e = 0; e < 32; ++e) {
      int idx = t * 32 + e;
      int i = idx >> 7;
      int j = idx & 127;
      tmp[e] = __expf(u.S[i * 132 + j] - mrow[i]);
    }
    __syncthreads();
#pragma unroll
    for (int e = 0; e < 32; ++e) {
      int idx = t * 32 + e;
      int i = idx >> 7;
      int j = idx & 127;
      u.P[i * 136 + j] = (_Float16)tmp[e];
    }
    __syncthreads();

    // ---- Oacc(i,c) += P(i,j) x Vt(j,c): A = P rm (LDS), B = v rm (global)
#pragma unroll
    for (int ks = 0; ks < 4; ++ks) {
      v16h af[4], bf[4];
#pragma unroll
      for (int ii = 0; ii < 4; ++ii)
        af[ii] = load_frag_rm(&u.P[(ii * 16) * 136 + ks * 32], 136);
#pragma unroll
      for (int ci = 0; ci < 4; ++ci)
        bf[ci] = load_frag_rm(&vb[(size_t)(wave * 64 + ci * 16) * N_DIM + j0 + ks * 32],
                              N_DIM);
#pragma unroll
      for (int ii = 0; ii < 4; ++ii)
#pragma unroll
        for (int ci = 0; ci < 4; ++ci)
          oacc[ii][ci] = wmma16(af[ii], bf[ci], oacc[ii][ci]);
    }
    __syncthreads();
  }

  const int ln = lane & 15;
  const int mh = (lane >> 4) * 8;
#pragma unroll
  for (int ii = 0; ii < 4; ++ii)
#pragma unroll
    for (int ci = 0; ci < 4; ++ci)
#pragma unroll
      for (int r = 0; r < 8; ++r) {
        int i = ii * 16 + mh + r;
        int c = wave * 64 + ci * 16 + ln;
        oT[((size_t)b * N_DIM + i0 + i) * C_DIM + c] = (_Float16)(oacc[ii][ci][r] * linv[i]);
      }
}

// ---------------------------------------------------------------- launch
extern "C" void kernel_launch(void* const* d_in, const int* in_sizes, int n_in,
                              void* d_out, int out_size, void* d_ws, size_t ws_size,
                              hipStream_t stream) {
  (void)in_sizes; (void)n_in; (void)out_size; (void)ws_size;
  const float* x  = (const float*)d_in[0];
  const float* wq = (const float*)d_in[1];
  const float* bq = (const float*)d_in[2];
  const float* wk = (const float*)d_in[3];
  const float* bk = (const float*)d_in[4];
  const float* wv = (const float*)d_in[5];
  const float* bv = (const float*)d_in[6];
  const float* wp = (const float*)d_in[7];
  const float* bp = (const float*)d_in[8];
  float* out = (float*)d_out;

  char* ws = (char*)d_ws;
  size_t off = 0;
  const size_t plane = (size_t)B_DIM * C_DIM * N_DIM;  // 8M elements
  _Float16* w16  = (_Float16*)(ws + off); off += (size_t)4 * CC * 2;
  _Float16* hnT  = (_Float16*)(ws + off); off += plane * 2;
  _Float16* qT   = (_Float16*)(ws + off); off += plane * 2;
  _Float16* kT   = (_Float16*)(ws + off); off += plane * 2;
  _Float16* v16b = (_Float16*)(ws + off); off += plane * 2;
  _Float16* oT   = (_Float16*)(ws + off); off += plane * 2;
  float2* nstats = (float2*)(ws + off);   off += (size_t)B_DIM * C_DIM * sizeof(float2);
  float2* stats  = (float2*)(ws + off);   off += (size_t)B_DIM * N_DIM * sizeof(float2);

  cvt_w_kernel<<<(4 * CC) / 256, 256, 0, stream>>>(wq, wk, wv, wp, w16);
  in_stats_kernel<<<B_DIM * C_DIM, 256, 0, stream>>>(x, nstats);
  norm_t_kernel<<<dim3(N_DIM / 64, C_DIM / 64, B_DIM), 256, 0, stream>>>(x, nstats, hnT);
  // qT, kT: M = n (32 tiles of 128), N = o (2 tiles of 256), z = b*2+sel
  gemm_rr_kernel<0><<<dim3(2, 32, B_DIM * 2), 256, 0, stream>>>(
      w16, hnT, qT, kT, bq, bk, nullptr, nullptr);
  // v: M = c (4 tiles), N = n (16 tiles), z = b
  gemm_rr_kernel<2><<<dim3(16, 4, B_DIM), 256, 0, stream>>>(
      w16, hnT, v16b, nullptr, bv, nullptr, nullptr, nullptr);
  attn_stats_kernel<<<dim3(N_DIM / 64, B_DIM), 256, 0, stream>>>(qT, kT, stats);
  attn_out_kernel<<<dim3(N_DIM / 64, B_DIM), 256, 0, stream>>>(qT, kT, v16b, stats, oT);
  // proj + residual: M = c (4 tiles), N = n (16 tiles), z = b
  gemm_rr_kernel<3><<<dim3(16, 4, B_DIM), 256, 0, stream>>>(
      w16, oT, nullptr, nullptr, bp, nullptr, x, out);
}
